// MaskedCausalAttention_19404662243505
// MI455X (gfx1250) — compile-verified, matrix operand order checked by reading, not run on hardware
//
#include <hip/hip_runtime.h>
#include <hip/hip_bf16.h>
#include <cstdint>

// ---------------------------------------------------------------------------
// Masked causal attention (MGDT mask) for MI455X / gfx1250, wave32 + WMMA.
// Pipeline: qkv proj (bf16 WMMA, 128x128 block tiles) -> flash attention
// (bf16 WMMA, online softmax, double-buffered async-to-LDS K/V staging) ->
// output proj (bf16 WMMA, fp32 out).
// ---------------------------------------------------------------------------

typedef __bf16 bf16_t;
typedef __attribute__((ext_vector_type(16))) __bf16 v16bf;
typedef __attribute__((ext_vector_type(8)))  __bf16 v8bf;
typedef __attribute__((ext_vector_type(8)))  float  v8f;

#define B_SZ 2
#define T_SZ 2048
#define C_SZ 1024
#define N_HD 16
#define D_HD 64
#define BT_SZ (B_SZ * T_SZ)

#define LDS_PAD 8
#define XS_STR (32 + LDS_PAD)   // 40 elems = 80B rows (16B multiple)
#define WT_STR (32 + LDS_PAD)
#define KS_STR (64 + LDS_PAD)   // 72 elems = 144B rows
#define VT_STR (32 + LDS_PAD)
#define PS_STR (32 + LDS_PAD)

// ---- CDNA5 async-tensor path -----------------------------------------------
// Generic pointers to LDS carry the shared aperture in the high 32 bits and
// the LDS byte offset in the low 32 bits (apertures are 4GB-aligned), so the
// low word is exactly the VDST LDS address the async instruction wants.
__device__ inline uint32_t lds_off(const void* p) {
  return (uint32_t)(uintptr_t)p;
}

// GLOBAL_LOAD_ASYNC_TO_LDS_B128 (VGLOBAL op 98, GV mode): per-lane 16B
// memory -> LDS, tracked by ASYNCcnt, no VGPR round-trip.
__device__ inline void async_ld_b128(uint32_t lds_byte_addr, const void* gaddr) {
  asm volatile("global_load_async_to_lds_b128 %0, %1, off"
               :: "v"(lds_byte_addr), "v"(gaddr) : "memory");
}

__device__ inline void wait_async0() {
  asm volatile("s_wait_asynccnt 0x0" ::: "memory");
}

// ---- WMMA fragment helpers --------------------------------------------------
// 16x32 bf16 operand fragment from a row-major tile.
// Per ISA 16-bit A layout: lanes 0-15 hold row=lane, K {0..7, 16..23};
// lanes 16-31 hold row=lane-16, K {8..15, 24..31}.
__device__ inline v16bf load_frag(const bf16_t* __restrict__ base, int stride, int lane) {
  const int r  = lane & 15;
  const int k0 = (lane & 16) ? 8 : 0;
  const bf16_t* p = base + r * stride + k0;
  v8bf lo = *(const v8bf*)(p);
  v8bf hi = *(const v8bf*)(p + 16);
  v16bf f;
#pragma unroll
  for (int i = 0; i < 8; ++i) { f[i] = lo[i]; f[i + 8] = hi[i]; }
  return f;
}

__device__ inline v8f wmma_bf16(v16bf a, v16bf b, v8f c) {
  // 8 args: (neg_a, A, neg_b, B, c_mod, C, reuse_a, reuse_b)
  return __builtin_amdgcn_wmma_f32_16x16x32_bf16(false, a, false, b, (short)0, c,
                                                 false, false);
}

__device__ inline v8f v8f_zero() {
  v8f z;
#pragma unroll
  for (int i = 0; i < 8; ++i) z[i] = 0.0f;
  return z;
}

// Reductions across the 16-lane half (masks 1,2,4,8 never cross halves).
__device__ inline float half16_max(float v) {
#pragma unroll
  for (int m = 1; m <= 8; m <<= 1) v = fmaxf(v, __shfl_xor(v, m, 32));
  return v;
}
__device__ inline float half16_sum(float v) {
#pragma unroll
  for (int m = 1; m <= 8; m <<= 1) v += __shfl_xor(v, m, 32);
  return v;
}

// ---------------------------------------------------------------------------
// Kernel 1: projection  out_bf16[b,h,t,d] = scale * (X @ W + bias)
// X fp32 [BT, C], W fp32 [C, C]. Block = 256 thr (8 waves), block tile
// 128x128, each wave owns a 64x32 subtile: 8 WMMAs per K-step from
// 4 A-fragments x 2 B-fragments (operand reuse raises WMMA:LDS ratio).
// fp32 sources need VALU conversion, so staging stays on load/convert/store.
// ---------------------------------------------------------------------------
__global__ __launch_bounds__(256)
void proj_qkv_kernel(const float* __restrict__ X, const float* __restrict__ W,
                     const float* __restrict__ bias, bf16_t* __restrict__ out,
                     float scale) {
  __shared__ __align__(16) bf16_t Xs[128 * XS_STR];
  __shared__ __align__(16) bf16_t Wt[128 * WT_STR];   // [n][k] transposed

  const int tid  = threadIdx.x;
  const int lane = tid & 31;
  const int wave = tid >> 5;
  const int wm   = wave & 1;   // 64-row strip
  const int wn   = wave >> 1;  // 32-col strip (0..3)
  const int nbase = blockIdx.x * 128;
  const int mbase = blockIdx.y * 128;

  v8f acc[4][2];
#pragma unroll
  for (int i = 0; i < 4; ++i)
#pragma unroll
    for (int j = 0; j < 2; ++j) acc[i][j] = v8f_zero();

  for (int kc = 0; kc < C_SZ; kc += 32) {
    // ---- stage X tile 128x32 (fp32 -> bf16) ----
    {
      const int col  = (tid & 7) * 4;
      const int row0 = tid >> 3;  // 0..31
#pragma unroll
      for (int p = 0; p < 4; ++p) {
        const int row = row0 + p * 32;
        const float4 v = *(const float4*)(X + (size_t)(mbase + row) * C_SZ + kc + col);
        bf16_t* dst = Xs + row * XS_STR + col;
        dst[0] = (bf16_t)v.x; dst[1] = (bf16_t)v.y;
        dst[2] = (bf16_t)v.z; dst[3] = (bf16_t)v.w;
      }
      if (kc + 32 < C_SZ)  // prefetch next X tile (global_prefetch_b8)
        __builtin_prefetch(X + (size_t)(mbase + row0) * C_SZ + kc + 32 + col, 0, 3);
    }
    // ---- stage W tile 32x128 transposed -> Wt[n][k] ----
    {
      const int n0  = (tid & 31) * 4;  // 0..124
      const int kk0 = tid >> 5;        // 0..7
#pragma unroll
      for (int p = 0; p < 4; ++p) {
        const int kk = kk0 + p * 8;
        const float4 v = *(const float4*)(W + (size_t)(kc + kk) * C_SZ + nbase + n0);
        Wt[(n0 + 0) * WT_STR + kk] = (bf16_t)v.x;
        Wt[(n0 + 1) * WT_STR + kk] = (bf16_t)v.y;
        Wt[(n0 + 2) * WT_STR + kk] = (bf16_t)v.z;
        Wt[(n0 + 3) * WT_STR + kk] = (bf16_t)v.w;
      }
      if (kc + 32 < C_SZ)
        __builtin_prefetch(W + (size_t)(kc + 32 + kk0) * C_SZ + nbase + n0, 0, 3);
    }
    __syncthreads();

    v16bf a[4], b[2];
#pragma unroll
    for (int mi = 0; mi < 4; ++mi)
      a[mi] = load_frag(Xs + (wm * 64 + mi * 16) * XS_STR, XS_STR, lane);
#pragma unroll
    for (int ni = 0; ni < 2; ++ni)
      b[ni] = load_frag(Wt + (wn * 32 + ni * 16) * WT_STR, WT_STR, lane);
#pragma unroll
    for (int mi = 0; mi < 4; ++mi)
#pragma unroll
      for (int ni = 0; ni < 2; ++ni)
        acc[mi][ni] = wmma_bf16(a[mi], b[ni], acc[mi][ni]);
    __syncthreads();
  }

  // ---- epilogue: split-head bf16 store (Q pre-scaled by 1/sqrt(D)) ----
  const int hih = (lane >> 4) & 1;
  const int ncl = lane & 15;
#pragma unroll
  for (int mi = 0; mi < 4; ++mi)
#pragma unroll
    for (int ni = 0; ni < 2; ++ni)
#pragma unroll
      for (int j = 0; j < 8; ++j) {
        const int m = mbase + wm * 64 + mi * 16 + j + hih * 8;
        const int n = nbase + wn * 32 + ni * 16 + ncl;
        const float val = (acc[mi][ni][j] + bias[n]) * scale;
        const int bb = m >> 11, tt = m & (T_SZ - 1);
        const int hh = n >> 6,  dd = n & 63;
        out[(((size_t)bb * N_HD + hh) * T_SZ + tt) * D_HD + dd] = (bf16_t)val;
      }
}

// ---------------------------------------------------------------------------
// Kernel 2: flash attention per (batch-head, 64-query tile).
// Block = 128 thr (4 waves), wave handles 16 query rows.
// Mask: key <= q  AND  !(key >= 5 && key % 4 == 1)   (MGDT return columns)
// K/V tiles are DOUBLE BUFFERED: tile t+1 is issued (K via
// GLOBAL_LOAD_ASYNC_TO_LDS_B128 on ASYNCcnt, V via load+transpose) before
// the WMMA work on tile t, so HBM/L2 latency overlaps matrix compute.
// ---------------------------------------------------------------------------
__global__ __launch_bounds__(128)
void attn_kernel(const bf16_t* __restrict__ Q, const bf16_t* __restrict__ K,
                 const bf16_t* __restrict__ V, bf16_t* __restrict__ CTX) {
  __shared__ __align__(16) bf16_t Ks[2][32 * KS_STR];    // [buf][key][d]
  __shared__ __align__(16) bf16_t Vt[2][64 * VT_STR];    // [buf][d][key]
  __shared__ __align__(16) bf16_t Ps[4][16 * PS_STR];    // per-wave P staging

  const int tid  = threadIdx.x;
  const int lane = tid & 31;
  const int wave = tid >> 5;
  const int qtile = blockIdx.x;       // 64 query rows per block
  const int bh    = blockIdx.y;       // b * N_HD + h
  const int bb = bh / N_HD;
  const int hh = bh % N_HD;

  const bf16_t* Qh = Q + (size_t)bh * T_SZ * D_HD;
  const bf16_t* Kh = K + (size_t)bh * T_SZ * D_HD;
  const bf16_t* Vh = V + (size_t)bh * T_SZ * D_HD;

  const int qbase = qtile * 64;
  const int qrow0 = qbase + wave * 16;

  // Q fragments for this wave's 16 rows (Q already holds q/sqrt(D))
  const v16bf qa0 = load_frag(Qh + (size_t)qrow0 * D_HD,      D_HD, lane);  // d 0..31
  const v16bf qa1 = load_frag(Qh + (size_t)qrow0 * D_HD + 32, D_HD, lane);  // d 32..63

  const int hi8 = (lane & 16) ? 8 : 0;
  const int col = lane & 15;

  float rmax[8], rsum[8];
  v8f o[4];
#pragma unroll
  for (int j = 0; j < 8; ++j) { rmax[j] = -1e30f; rsum[j] = 0.0f; }
#pragma unroll
  for (int dg = 0; dg < 4; ++dg) o[dg] = v8f_zero();

  const int key0 = tid >> 3;         // 0..15
  const int d0   = (tid & 7) * 8;    // element offset, 16B chunks

  // Stage one 32-key tile into buffer `buf`.
  auto stage = [&](int kt, int buf) {
    const int kb = kt * 32;
    const uint32_t ksB = lds_off(&Ks[buf][0]);
#pragma unroll
    for (int p = 0; p < 2; ++p) {
      const int key = key0 + p * 16;
      async_ld_b128(ksB + (uint32_t)(key * KS_STR + d0) * 2,
                    Kh + (size_t)(kb + key) * D_HD + d0);
      const v8bf vv = *(const v8bf*)(Vh + (size_t)(kb + key) * D_HD + d0);
#pragma unroll
      for (int i = 0; i < 8; ++i) Vt[buf][(d0 + i) * VT_STR + key] = vv[i];
    }
  };

  const int ntiles = (qbase + 64) / 32;  // causal bound, 32 keys per tile
  stage(0, 0);
  wait_async0();
  __syncthreads();

  for (int kt = 0; kt < ntiles; ++kt) {
    const int cur = kt & 1;
    const int kb = kt * 32;
    // ---- issue next tile into the other buffer before computing this one ----
    if (kt + 1 < ntiles) stage(kt + 1, cur ^ 1);
    if (kt + 2 < ntiles) {  // L2 prefetch two tiles ahead
      __builtin_prefetch(Kh + (size_t)(kb + 64 + key0) * D_HD + d0, 0, 3);
      __builtin_prefetch(Vh + (size_t)(kb + 64 + key0) * D_HD + d0, 0, 3);
    }

    // ---- S = Q K^T  (two 16x16 tiles, K-dim = D = 64 -> 2 WMMAs each) ----
    v8f s[2];
#pragma unroll
    for (int g = 0; g < 2; ++g) {
      const v16bf kb0 = load_frag(Ks[cur] + (g * 16) * KS_STR,      KS_STR, lane);
      const v16bf kb1 = load_frag(Ks[cur] + (g * 16) * KS_STR + 32, KS_STR, lane);
      s[g] = wmma_bf16(qa1, kb1, wmma_bf16(qa0, kb0, v8f_zero()));
    }

    // ---- mask ----
#pragma unroll
    for (int g = 0; g < 2; ++g)
#pragma unroll
      for (int j = 0; j < 8; ++j) {
        const int q = qrow0 + j + hi8;
        const int kc = kb + g * 16 + col;
        const bool ok = (kc <= q) && !((kc >= 5) && ((kc & 3) == 1));
        if (!ok) s[g][j] = -1e30f;
      }

    // ---- online softmax update ----
    float rescale[8];
#pragma unroll
    for (int j = 0; j < 8; ++j) {
      const float tmax = half16_max(fmaxf(s[0][j], s[1][j]));
      const float nmax = fmaxf(rmax[j], tmax);
      rescale[j] = __expf(rmax[j] - nmax);
      rmax[j] = nmax;
    }
#pragma unroll
    for (int g = 0; g < 2; ++g)
#pragma unroll
      for (int j = 0; j < 8; ++j) s[g][j] = __expf(s[g][j] - rmax[j]);
#pragma unroll
    for (int j = 0; j < 8; ++j)
      rsum[j] = rsum[j] * rescale[j] + half16_sum(s[0][j] + s[1][j]);
#pragma unroll
    for (int dg = 0; dg < 4; ++dg)
#pragma unroll
      for (int j = 0; j < 8; ++j) o[dg][j] *= rescale[j];

    // ---- re-stage P through LDS into A-fragment layout (intra-wave,
    //      LDS ops from one wave are in-order so no barrier needed) ----
    bf16_t* ps = &Ps[wave][0];
#pragma unroll
    for (int g = 0; g < 2; ++g)
#pragma unroll
      for (int j = 0; j < 8; ++j)
        ps[(j + hi8) * PS_STR + g * 16 + col] = (bf16_t)s[g][j];
    const v16bf pa = load_frag(ps, PS_STR, lane);

    // ---- O += P @ V ----
#pragma unroll
    for (int dg = 0; dg < 4; ++dg) {
      const v16bf vb = load_frag(Vt[cur] + (dg * 16) * VT_STR, VT_STR, lane);
      o[dg] = wmma_bf16(pa, vb, o[dg]);
    }

    // ---- tile t+1 (async K + manual V) must land before next iteration ----
    wait_async0();
    __syncthreads();
  }

  // ---- normalize + store ctx as bf16 [b, t, h*64 + d] ----
#pragma unroll
  for (int dg = 0; dg < 4; ++dg)
#pragma unroll
    for (int j = 0; j < 8; ++j) {
      const int t = qrow0 + j + hi8;
      const int c = hh * D_HD + dg * 16 + col;
      const float v = o[dg][j] / fmaxf(rsum[j], 1e-30f);
      CTX[((size_t)bb * T_SZ + t) * C_SZ + c] = (bf16_t)v;
    }
}

// ---------------------------------------------------------------------------
// Kernel 3: output projection  out_f32 = ctx(bf16) @ Wp + bp
// Same 128x128 block tiling; ctx is already bf16 in the LDS layout, so its
// staging uses the async global-to-LDS path.
// ---------------------------------------------------------------------------
__global__ __launch_bounds__(256)
void out_proj_kernel(const bf16_t* __restrict__ Xin, const float* __restrict__ W,
                     const float* __restrict__ bias, float* __restrict__ out) {
  __shared__ __align__(16) bf16_t Xs[128 * XS_STR];
  __shared__ __align__(16) bf16_t Wt[128 * WT_STR];

  const int tid  = threadIdx.x;
  const int lane = tid & 31;
  const int wave = tid >> 5;
  const int wm   = wave & 1;
  const int wn   = wave >> 1;
  const int nbase = blockIdx.x * 128;
  const int mbase = blockIdx.y * 128;
  const uint32_t xsBase = lds_off(Xs);

  v8f acc[4][2];
#pragma unroll
  for (int i = 0; i < 4; ++i)
#pragma unroll
    for (int j = 0; j < 2; ++j) acc[i][j] = v8f_zero();

  for (int kc = 0; kc < C_SZ; kc += 32) {
    // ---- ctx tile 128x32 bf16: async global -> LDS ----
    {
      const int c0   = (tid & 3) * 8;
      const int row0 = tid >> 2;  // 0..63
#pragma unroll
      for (int p = 0; p < 2; ++p) {
        const int row = row0 + p * 64;
        async_ld_b128(xsBase + (uint32_t)(row * XS_STR + c0) * 2,
                      Xin + (size_t)(mbase + row) * C_SZ + kc + c0);
      }
      if (kc + 32 < C_SZ)
        __builtin_prefetch(Xin + (size_t)(mbase + row0) * C_SZ + kc + 32 + c0, 0, 3);
    }
    // ---- W tile 32x128 fp32 -> bf16, transposed -> Wt[n][k] ----
    {
      const int n0  = (tid & 31) * 4;
      const int kk0 = tid >> 5;
#pragma unroll
      for (int p = 0; p < 4; ++p) {
        const int kk = kk0 + p * 8;
        const float4 v = *(const float4*)(W + (size_t)(kc + kk) * C_SZ + nbase + n0);
        Wt[(n0 + 0) * WT_STR + kk] = (bf16_t)v.x;
        Wt[(n0 + 1) * WT_STR + kk] = (bf16_t)v.y;
        Wt[(n0 + 2) * WT_STR + kk] = (bf16_t)v.z;
        Wt[(n0 + 3) * WT_STR + kk] = (bf16_t)v.w;
      }
      if (kc + 32 < C_SZ)
        __builtin_prefetch(W + (size_t)(kc + 32 + kk0) * C_SZ + nbase + n0, 0, 3);
    }
    wait_async0();
    __syncthreads();

    v16bf a[4], b[2];
#pragma unroll
    for (int mi = 0; mi < 4; ++mi)
      a[mi] = load_frag(Xs + (wm * 64 + mi * 16) * XS_STR, XS_STR, lane);
#pragma unroll
    for (int ni = 0; ni < 2; ++ni)
      b[ni] = load_frag(Wt + (wn * 32 + ni * 16) * WT_STR, WT_STR, lane);
#pragma unroll
    for (int mi = 0; mi < 4; ++mi)
#pragma unroll
      for (int ni = 0; ni < 2; ++ni)
        acc[mi][ni] = wmma_bf16(a[mi], b[ni], acc[mi][ni]);
    __syncthreads();
  }

  const int hih = (lane >> 4) & 1;
  const int ncl = lane & 15;
#pragma unroll
  for (int mi = 0; mi < 4; ++mi)
#pragma unroll
    for (int ni = 0; ni < 2; ++ni)
#pragma unroll
      for (int j = 0; j < 8; ++j) {
        const int m = mbase + wm * 64 + mi * 16 + j + hih * 8;
        const int n = nbase + wn * 32 + ni * 16 + ncl;
        out[(size_t)m * C_SZ + n] = acc[mi][ni][j] + bias[n];
      }
}

// ---------------------------------------------------------------------------
// Launch: proj Q/K/V -> flash attention -> output projection.
// Workspace layout (bf16): Q | K | V | CTX  = 4 * 8 MB = 32 MB.
// ---------------------------------------------------------------------------
extern "C" void kernel_launch(void* const* d_in, const int* in_sizes, int n_in,
                              void* d_out, int out_size, void* d_ws, size_t ws_size,
                              hipStream_t stream) {
  (void)in_sizes; (void)n_in; (void)out_size; (void)ws_size;
  const float* x  = (const float*)d_in[0];
  const float* Wq = (const float*)d_in[1];
  const float* bq = (const float*)d_in[2];
  const float* Wk = (const float*)d_in[3];
  const float* bk = (const float*)d_in[4];
  const float* Wv = (const float*)d_in[5];
  const float* bv = (const float*)d_in[6];
  const float* Wp = (const float*)d_in[7];
  const float* bp = (const float*)d_in[8];
  // d_in[9] = mask (computed analytically in-kernel)

  const size_t headElems = (size_t)B_SZ * N_HD * T_SZ * D_HD;  // == B*T*C
  bf16_t* Qb  = (bf16_t*)d_ws;
  bf16_t* Kb  = Qb + headElems;
  bf16_t* Vb  = Kb + headElems;
  bf16_t* CTX = Vb + headElems;

  const dim3 gp(C_SZ / 128, BT_SZ / 128);  // 8 x 32
  proj_qkv_kernel<<<gp, 256, 0, stream>>>(x, Wq, bq, Qb, 0.125f);  // 1/sqrt(64)
  proj_qkv_kernel<<<gp, 256, 0, stream>>>(x, Wk, bk, Kb, 1.0f);
  proj_qkv_kernel<<<gp, 256, 0, stream>>>(x, Wv, bv, Vb, 1.0f);

  attn_kernel<<<dim3(T_SZ / 64, B_SZ * N_HD), 128, 0, stream>>>(Qb, Kb, Vb, CTX);

  out_proj_kernel<<<gp, 256, 0, stream>>>(CTX, Wp, bp, (float*)d_out);
}
